// CausalSelfAttention_39737037423076
// MI455X (gfx1250) — compile-verified, matrix operand order checked by reading, not run on hardware
//
#include <hip/hip_runtime.h>
#include <hip/hip_bf16.h>
#include <math.h>

// ---------------------------------------------------------------------------
// CDNA5 (gfx1250, wave32) causal self-attention:
//   q,k,v = x @ W{q,k,v}^T  (WMMA f16 -> f32, weights staged to LDS via TDM)
//   flash-attention per (b,h) with online softmax (WMMA)
//   out = attn_out @ W_o^T  (WMMA, f32 out)
// ---------------------------------------------------------------------------

typedef __attribute__((ext_vector_type(16))) _Float16 v16h;
typedef __attribute__((ext_vector_type(8)))  _Float16 h8;
typedef __attribute__((ext_vector_type(8)))  float    v8f;
typedef __attribute__((ext_vector_type(4)))  unsigned int v4u;
typedef __attribute__((ext_vector_type(8)))  int      v8i;
typedef __attribute__((ext_vector_type(4)))  int      v4i;

static constexpr int B_  = 4;
static constexpr int T_  = 2048;
static constexpr int C_  = 1024;
static constexpr int NH_ = 16;
static constexpr int HD_ = 64;

#define DEVINL __device__ __forceinline__

union FragU { v16h v; h8 h[2]; };

// Load a 16x32 f16 operand fragment (A- and B-matrix layouts are symmetric on
// CDNA5): lane r=lane&15 is the M (or N) row; half-wave selects K base 0/8;
// per lane two contiguous 8-half chunks at K=kb and K=kb+16.
DEVINL v16h load_frag(const _Float16* __restrict__ base, int stride, int lane) {
  const int r  = lane & 15;
  const int kb = (lane >> 4) << 3;   // 0 or 8
  const _Float16* p = base + r * stride + kb;
  FragU u;
  u.h[0] = *(const h8*)(p);
  u.h[1] = *(const h8*)(p + 16);
  return u.v;
}

DEVINL v8f wmma_f16(v16h a, v16h b, v8f c) {
  return __builtin_amdgcn_wmma_f32_16x16x32_f16(false, a, false, b,
                                                (short)0, c, false, false);
}

DEVINL v8f zero8() {
  v8f z;
#pragma unroll
  for (int i = 0; i < 8; ++i) z[i] = 0.0f;
  return z;
}

// ---------------------------------------------------------------------------
__global__ void cvt_kernel(const float* __restrict__ s, _Float16* __restrict__ d,
                           int nElem) {
  int i  = blockIdx.x * blockDim.x + threadIdx.x;
  int st = gridDim.x * blockDim.x;
  for (; i < nElem; i += st) d[i] = (_Float16)s[i];
}

// ---------------------------------------------------------------------------
// C[m,n] = sum_k A[m,k] * W[n,k].  A:[M,K] f16 row-major, W:[N,K] f16 row-major.
// mode 0: f32 output, row-major [M,N];  mode 1: f16 output, head-split.
// Block: 256 threads = 8 waves (4x2), 128x128 tile, wave tile 32x64.
// Weight slab (128x32) staged into padded LDS by the Tensor Data Mover when
// available (pad_enable converts the 64B rows into stride-80B LDS rows).
__global__ __launch_bounds__(256)
void gemm_wmma_kernel(const _Float16* __restrict__ A,
                      const _Float16* __restrict__ W,
                      float* __restrict__ outF,
                      _Float16* __restrict__ outH,
                      int M, int N, int K, int mode) {
  constexpr int KSTR = 40;  // padded LDS row stride (halves): 64B row + 16B pad
  __shared__ alignas(16) _Float16 As[128 * KSTR];
  __shared__ alignas(16) _Float16 Bs[128 * KSTR];

  const int tid  = threadIdx.x;
  const int lane = tid & 31;
  const int wave = tid >> 5;
  const int m0 = blockIdx.y * 128;
  const int n0 = blockIdx.x * 128;
  const int wm = (wave >> 1) * 32;   // 0,32,64,96
  const int wn = (wave & 1) * 64;    // 0,64

  v8f acc[2][4];
#pragma unroll
  for (int i = 0; i < 2; ++i)
#pragma unroll
    for (int j = 0; j < 4; ++j) acc[i][j] = zero8();

  for (int kb = 0; kb < K; kb += 32) {
#if __has_builtin(__builtin_amdgcn_tensor_load_to_lds)
    // --- TDM path for the weight slab: 128 rows x 32 halves, padded into LDS.
    if (wave == 0) {
      const unsigned long long gaddr =
          (unsigned long long)(const void*)&W[(size_t)n0 * K + kb];
      const unsigned int laddr = (unsigned int)(size_t)(void*)&Bs[0];
      v4u g0;
      g0[0] = 1u;                                   // count=1 (valid descriptor)
      g0[1] = laddr;                                // lds_addr
      g0[2] = (unsigned int)gaddr;                  // global_addr[31:0]
      g0[3] = ((unsigned int)(gaddr >> 32) & 0x01FFFFFFu) | (2u << 30); // type=2
      v8i g1;
      g1[0] = (int)((1u << 16)      // data_size = 1 (2 bytes)
                  | (1u << 20)      // pad_enable
                  | (3u << 22)      // pad_interval: 16 dwords (one 64B row)
                  | (3u << 25));    // pad_amount: 4 dwords (16B)
      const unsigned int d0 = (unsigned int)(K - kb);   // tensor_dim0 (elems)
      const unsigned int d1 = (unsigned int)(N - n0);   // tensor_dim1 (rows)
      g1[1] = (int)((d0 & 0xFFFFu) << 16);              // dim0[15:0]
      g1[2] = (int)((d0 >> 16) | ((d1 & 0xFFFFu) << 16));  // dim0[31:16] | dim1[15:0]
      g1[3] = (int)((d1 >> 16) | (32u << 16));          // dim1[31:16] | tile_dim0=32
      g1[4] = 128;                                      // tile_dim1=128
      g1[5] = K;                                        // tensor_dim0_stride[31:0]
      g1[6] = 0;
      g1[7] = 0;
      v4i z4; z4[0] = 0; z4[1] = 0; z4[2] = 0; z4[3] = 0;
#if defined(__clang_major__) && __clang_major__ >= 23
      v8i z8;
#pragma unroll
      for (int q = 0; q < 8; ++q) z8[q] = 0;
      __builtin_amdgcn_tensor_load_to_lds(g0, g1, z4, z4, z8, 0);
#else
      __builtin_amdgcn_tensor_load_to_lds(g0, g1, z4, z4, 0);
#endif
    }
#endif

    // --- cooperative vector-load path for the A slab (128 x 32 halves)
#pragma unroll
    for (int i = 0; i < 2; ++i) {
      int c   = tid + i * 256;      // 0..511
      int row = c >> 2;             // 0..127
      int off = (c & 3) * 8;        // 0,8,16,24
      *(h8*)&As[row * KSTR + off] =
          *(const h8*)&A[(size_t)(m0 + row) * K + kb + off];
#if !__has_builtin(__builtin_amdgcn_tensor_load_to_lds)
      *(h8*)&Bs[row * KSTR + off] =
          *(const h8*)&W[(size_t)(n0 + row) * K + kb + off];
#endif
    }
    if (kb + 32 < K)
      __builtin_prefetch(&A[(size_t)(m0 + (tid >> 1)) * K + kb + 32], 0, 0);

#if __has_builtin(__builtin_amdgcn_tensor_load_to_lds)
    if (wave == 0) __builtin_amdgcn_s_wait_tensorcnt(0);
#endif
    __syncthreads();

    v16h a[2], b[4];
#pragma unroll
    for (int ti = 0; ti < 2; ++ti)
      a[ti] = load_frag(&As[(wm + 16 * ti) * KSTR], KSTR, lane);
#pragma unroll
    for (int tj = 0; tj < 4; ++tj)
      b[tj] = load_frag(&Bs[(wn + 16 * tj) * KSTR], KSTR, lane);
#pragma unroll
    for (int ti = 0; ti < 2; ++ti)
#pragma unroll
      for (int tj = 0; tj < 4; ++tj)
        acc[ti][tj] = wmma_f16(a[ti], b[tj], acc[ti][tj]);
    __syncthreads();
  }

  // C/D layout: lanes 0-15 hold rows r, lanes 16-31 rows r+8; N = lane&15.
  const int n  = lane & 15;
  const int mh = (lane >> 4) * 8;
#pragma unroll
  for (int ti = 0; ti < 2; ++ti)
#pragma unroll
    for (int tj = 0; tj < 4; ++tj) {
      const int col = n0 + wn + tj * 16 + n;
#pragma unroll
      for (int r = 0; r < 8; ++r) {
        const int row = m0 + wm + ti * 16 + mh + r;
        const float vv = acc[ti][tj][r];
        if (mode == 0) {
          outF[(size_t)row * N + col] = vv;
        } else {
          const int bb = row / T_, tt = row % T_;
          const int hh = col / HD_, dd = col % HD_;
          outH[(((size_t)bb * NH_ + hh) * T_ + tt) * HD_ + dd] = (_Float16)vv;
        }
      }
    }
}

// ---------------------------------------------------------------------------
// Flash attention. Grid: (T/64, B*NH). Block: 128 threads = 4 waves; wave w
// owns q-rows [qB + 16w, qB + 16w + 16). Q/K/V are f16 in [B,NH,T,HD].
// Output O is f16 in [B*T, C] (row = b*T + t, col = h*HD + d).
__global__ __launch_bounds__(128)
void attn_kernel(const _Float16* __restrict__ Q,
                 const _Float16* __restrict__ Km,
                 const _Float16* __restrict__ V,
                 _Float16* __restrict__ O) {
  constexpr int KSTR = 72;  // K tile: 32 keys x 64 dims (padded)
  constexpr int VSTR = 40;  // V^T tile: 64 dims x 32 keys (padded)
  constexpr int PSTR = 40;  // P staging: per-wave 16 x 32 (padded)
  __shared__ alignas(16) _Float16 Ks[32 * KSTR];
  __shared__ alignas(16) _Float16 Vt[64 * VSTR];
  __shared__ alignas(16) _Float16 Ps[4 * 16 * PSTR];

  const int tid  = threadIdx.x;
  const int lane = tid & 31;
  const int wave = tid >> 5;
  const int bh = blockIdx.y;          // b*NH + h
  const int bb = bh >> 4;             // NH_ == 16
  const int hh = bh & 15;
  const int qB = blockIdx.x * 64;
  const int q0 = qB + wave * 16;

  const _Float16* Qb = Q  + (size_t)bh * T_ * HD_;
  const _Float16* Kb = Km + (size_t)bh * T_ * HD_;
  const _Float16* Vb = V  + (size_t)bh * T_ * HD_;

  v16h aq0 = load_frag(Qb + (size_t)q0 * HD_ +  0, HD_, lane);
  v16h aq1 = load_frag(Qb + (size_t)q0 * HD_ + 32, HD_, lane);

  v8f o[4];
#pragma unroll
  for (int nt = 0; nt < 4; ++nt) o[nt] = zero8();
  float rmax[8], rsum[8];
#pragma unroll
  for (int r = 0; r < 8; ++r) { rmax[r] = -1e30f; rsum[r] = 0.0f; }

  const int n  = lane & 15;
  const int mh = (lane >> 4) * 8;
  const float scale = 0.125f;          // 1/sqrt(64)
  _Float16* Pw = &Ps[wave * 16 * PSTR];

  const int kEnd = qB + 64;            // causal: no keys beyond block's last q
  for (int k0 = 0; k0 < kEnd; k0 += 32) {
#pragma unroll
    for (int i = 0; i < 2; ++i) {
      int c   = tid + i * 128;   // 0..255
      int kk  = c >> 3;          // 0..31
      int off = (c & 7) * 8;     // 0..56
      *(h8*)&Ks[kk * KSTR + off] =
          *(const h8*)&Kb[(size_t)(k0 + kk) * HD_ + off];
      h8 vv = *(const h8*)&Vb[(size_t)(k0 + kk) * HD_ + off];
#pragma unroll
      for (int j = 0; j < 8; ++j) Vt[(off + j) * VSTR + kk] = vv[j];
    }
    __syncthreads();

    if (k0 <= q0 + 15) {  // wave-uniform: EXEC stays all-ones for WMMA
      v8f s0 = zero8(), s1 = zero8();
      v16h bk0_0 = load_frag(&Ks[ 0 * KSTR +  0], KSTR, lane);
      v16h bk0_1 = load_frag(&Ks[ 0 * KSTR + 32], KSTR, lane);
      v16h bk1_0 = load_frag(&Ks[16 * KSTR +  0], KSTR, lane);
      v16h bk1_1 = load_frag(&Ks[16 * KSTR + 32], KSTR, lane);
      s0 = wmma_f16(aq0, bk0_0, s0);
      s0 = wmma_f16(aq1, bk0_1, s0);
      s1 = wmma_f16(aq0, bk1_0, s1);
      s1 = wmma_f16(aq1, bk1_1, s1);

#pragma unroll
      for (int r = 0; r < 8; ++r) {
        const int qi = q0 + mh + r;
        float v0 = s0[r] * scale;
        float v1 = s1[r] * scale;
        v0 = (k0 + n      <= qi) ? v0 : -1e30f;
        v1 = (k0 + 16 + n <= qi) ? v1 : -1e30f;
        float m = fmaxf(v0, v1);
        m = fmaxf(m, __shfl_xor(m, 1, 32));
        m = fmaxf(m, __shfl_xor(m, 2, 32));
        m = fmaxf(m, __shfl_xor(m, 4, 32));
        m = fmaxf(m, __shfl_xor(m, 8, 32));
        const float nm    = fmaxf(rmax[r], m);
        const float alpha = __expf(rmax[r] - nm);
        const float p0 = __expf(v0 - nm);
        const float p1 = __expf(v1 - nm);
        float ts = p0 + p1;
        ts += __shfl_xor(ts, 1, 32);
        ts += __shfl_xor(ts, 2, 32);
        ts += __shfl_xor(ts, 4, 32);
        ts += __shfl_xor(ts, 8, 32);
        rsum[r] = rsum[r] * alpha + ts;
        rmax[r] = nm;
#pragma unroll
        for (int nt = 0; nt < 4; ++nt) o[nt][r] *= alpha;
        Pw[(mh + r) * PSTR + n]      = (_Float16)p0;
        Pw[(mh + r) * PSTR + 16 + n] = (_Float16)p1;
      }

      v16h ap = load_frag(Pw, PSTR, lane);
#pragma unroll
      for (int nt = 0; nt < 4; ++nt) {
        v16h bv = load_frag(&Vt[(nt * 16) * VSTR], VSTR, lane);
        o[nt] = wmma_f16(ap, bv, o[nt]);
      }
    }
    __syncthreads();
  }

#pragma unroll
  for (int r = 0; r < 8; ++r) {
    const float inv = 1.0f / rsum[r];
    const int qt = q0 + mh + r;
    const size_t base = ((size_t)bb * T_ + qt) * C_ + (size_t)hh * HD_;
#pragma unroll
    for (int nt = 0; nt < 4; ++nt)
      O[base + nt * 16 + n] = (_Float16)(o[nt][r] * inv);
  }
}

// ---------------------------------------------------------------------------
extern "C" void kernel_launch(void* const* d_in, const int* in_sizes, int n_in,
                              void* d_out, int out_size, void* d_ws, size_t ws_size,
                              hipStream_t stream) {
  (void)in_sizes; (void)n_in; (void)out_size; (void)ws_size;

  const float* x  = (const float*)d_in[0];
  const float* Wq = (const float*)d_in[1];
  const float* Wk = (const float*)d_in[2];
  const float* Wv = (const float*)d_in[3];
  const float* Wo = (const float*)d_in[4];
  float* out = (float*)d_out;

  char* ws = (char*)d_ws;
  const size_t MB = 1ull << 20;
  _Float16* xh  = (_Float16*)(ws +  0 * MB);  // 16 MB: x as f16 [M, C]
  _Float16* wqh = (_Float16*)(ws + 16 * MB);  //  2 MB each
  _Float16* wkh = (_Float16*)(ws + 18 * MB);
  _Float16* wvh = (_Float16*)(ws + 20 * MB);
  _Float16* woh = (_Float16*)(ws + 22 * MB);
  _Float16* Qh  = (_Float16*)(ws + 24 * MB);  // 16 MB each, [B,NH,T,HD]
  _Float16* Kh  = (_Float16*)(ws + 40 * MB);
  _Float16* Vh  = (_Float16*)(ws + 56 * MB);
  _Float16* Oh  = xh;  // reuse x's region after projections complete

  const int M = B_ * T_;  // 8192

  cvt_kernel<<<dim3(1024), dim3(256), 0, stream>>>(x,  xh,  M * C_);
  cvt_kernel<<<dim3(256),  dim3(256), 0, stream>>>(Wq, wqh, C_ * C_);
  cvt_kernel<<<dim3(256),  dim3(256), 0, stream>>>(Wk, wkh, C_ * C_);
  cvt_kernel<<<dim3(256),  dim3(256), 0, stream>>>(Wv, wvh, C_ * C_);
  cvt_kernel<<<dim3(256),  dim3(256), 0, stream>>>(Wo, woh, C_ * C_);

  dim3 gg(C_ / 128, M / 128);   // (8, 64)
  gemm_wmma_kernel<<<gg, 256, 0, stream>>>(xh, wqh, nullptr, Qh, M, C_, C_, 1);
  gemm_wmma_kernel<<<gg, 256, 0, stream>>>(xh, wkh, nullptr, Kh, M, C_, C_, 1);
  gemm_wmma_kernel<<<gg, 256, 0, stream>>>(xh, wvh, nullptr, Vh, M, C_, C_, 1);

  attn_kernel<<<dim3(T_ / 64, B_ * NH_), 128, 0, stream>>>(Qh, Kh, Vh, Oh);

  gemm_wmma_kernel<<<gg, 256, 0, stream>>>(Oh, woh, out, nullptr, M, C_, C_, 0);
}
